// SimpleDeepModel_66211215835467
// MI455X (gfx1250) — compile-verified, matrix operand order checked by reading, not run on hardware
//
#include <hip/hip_runtime.h>
#include <hip/hip_bf16.h>

#define T_SEQ 4096
#define EMB   300
#define EMBP  320    // K padded to multiple of 32 for wmma
#define HID   500
#define HIDP  512    // K padded
#define GATES 2000   // 4*HID
#define NCLS  90
#define NWG_REC 25   // 25 blocks * 8 waves = 200 waves (100 per layer)
#define REC_THREADS 256

typedef __attribute__((ext_vector_type(16))) __bf16 v16bf;
typedef __attribute__((ext_vector_type(8)))  __bf16 bf16x8;
typedef __attribute__((ext_vector_type(8)))  float  v8f;

// ---------------- embedding gather -> padded bf16 [T, EMBP] ----------------
__global__ void k_embed(const int* __restrict__ seq,
                        const float* __restrict__ emb,
                        __bf16* __restrict__ xe) {
    int idx = blockIdx.x * blockDim.x + threadIdx.x;
    if (idx >= T_SEQ * EMBP) return;
    int t = idx / EMBP, e = idx - t * EMBP;
    float v = (e < EMB) ? emb[(size_t)seq[t] * EMB + e] : 0.0f;
    xe[idx] = (__bf16)v;
}

// ---------------- f32 weights -> padded bf16 ----------------
__global__ void k_cvt(const float* __restrict__ src, __bf16* __restrict__ dst,
                      int rows, int sk, int dk) {
    int idx = blockIdx.x * blockDim.x + threadIdx.x;
    if (idx >= rows * dk) return;
    int r = idx / dk, c = idx - r * dk;
    dst[idx] = (__bf16)((c < sk) ? src[(size_t)r * sk + c] : 0.0f);
}

// ---------------- init small state buffers ----------------
__global__ void k_init(__bf16* h1buf, __bf16* h2buf, float* h2last, int* syncc) {
    int i = threadIdx.x;
    if (i < 2 * HIDP) { h1buf[i] = (__bf16)0.0f; h2buf[i] = (__bf16)0.0f; }
    if (i < HIDP) h2last[i] = 0.0f;
    if (i == 0) *syncc = 0;
}

// ---------------- WMMA GEMM: xp[T,GATES] = xe[T,EMBP] @ Wih0^T + (b_ih0+b_hh0) ----
// one wave per 16x16 output tile; A and B fragments loaded as two b128 per lane
// following the documented 16-bit A/B VGPR striping (two contiguous 8-elem K runs).
__global__ void __launch_bounds__(256)
k_gemm_xproj(const __bf16* __restrict__ xe, const __bf16* __restrict__ wb,
             const float* __restrict__ bi, const float* __restrict__ bh,
             float* __restrict__ xp) {
    int wave = threadIdx.x >> 5;
    int lane = threadIdx.x & 31;
    int m0 = blockIdx.y * 16;
    int n0 = (blockIdx.x * 8 + wave) * 16;
    if (n0 >= GATES) return;
    int half = lane >> 4;      // selects K sub-run per ISA layout
    int l16  = lane & 15;      // M row for A, N col for B

    const __bf16* arow = xe + (size_t)(m0 + l16) * EMBP + half * 8;
    const __bf16* brow = wb + (size_t)(n0 + l16) * EMBP + half * 8;

    v8f acc = {};
#pragma unroll
    for (int kk = 0; kk < EMBP; kk += 32) {
        bf16x8 alo = *(const bf16x8*)(arow + kk);
        bf16x8 ahi = *(const bf16x8*)(arow + kk + 16);
        bf16x8 blo = *(const bf16x8*)(brow + kk);
        bf16x8 bhi = *(const bf16x8*)(brow + kk + 16);
        if (kk + 32 < EMBP) __builtin_prefetch(brow + kk + 32, 0, 1);
        v16bf a, b;
#pragma unroll
        for (int e = 0; e < 8; ++e) {
            a[e] = alo[e]; a[e + 8] = ahi[e];
            b[e] = blo[e]; b[e + 8] = bhi[e];
        }
        acc = __builtin_amdgcn_wmma_f32_16x16x32_bf16(
            /*neg_a=*/false, a, /*neg_b=*/false, b,
            /*c_mod=*/(short)0, acc, /*reuse_a=*/false, /*reuse_b=*/false);
    }
    // C/D layout: VGPR r -> M = r + half*8 ; N = lane&15
    int n = n0 + l16;
    float bias = bi[n] + bh[n];
#pragma unroll
    for (int r = 0; r < 8; ++r) {
        int m = m0 + r + half * 8;
        xp[(size_t)m * GATES + n] = acc[r] + bias;
    }
}

// ---------------- recurrence helpers ----------------
__device__ __forceinline__ void load_xvec(const __bf16* __restrict__ x, int lane,
                                          float* xl) {
    const bf16x8* xp = (const bf16x8*)(x + lane * 16);
    bf16x8 x0 = xp[0], x1 = xp[1];
#pragma unroll
    for (int e = 0; e < 8; ++e) { xl[e] = (float)x0[e]; xl[e + 8] = (float)x1[e]; }
}

// full-row dot of length 512: lane l covers elements [16l, 16l+16), xor-reduce
__device__ __forceinline__ float wave_dot512(const __bf16* __restrict__ wrow, int lane,
                                             const float* xl) {
    const bf16x8* wp = (const bf16x8*)(wrow + lane * 16);
    bf16x8 w0 = wp[0], w1 = wp[1];
    float acc = 0.0f;
#pragma unroll
    for (int e = 0; e < 8; ++e)
        acc += (float)w0[e] * xl[e] + (float)w1[e] * xl[e + 8];
#pragma unroll
    for (int off = 16; off; off >>= 1) acc += __shfl_xor(acc, off, 32);
    return acc;
}

__device__ __forceinline__ float sigmoidf_(float x) {
    return 1.0f / (1.0f + __expf(-x));
}

// ---------------- persistent two-layer recurrence, one grid sync per step ----
// waves 0..99  : layer0, hidden indices j = gw + {0,100,200,300,400}
// waves 100..199: layer1 (skewed one step behind), same j partition
__global__ void __launch_bounds__(REC_THREADS)
k_recur(const float* __restrict__ xp0,
        const __bf16* __restrict__ whh0, const __bf16* __restrict__ wih1,
        const __bf16* __restrict__ whh1,
        const float* __restrict__ bih1, const float* __restrict__ bhh1,
        __bf16* __restrict__ h1buf, __bf16* __restrict__ h2buf,
        float* __restrict__ h2last, int* syncc) {
    int wid = threadIdx.x >> 5, lane = threadIdx.x & 31;
    int gw = blockIdx.x * (REC_THREADS / 32) + wid;
    bool isL0 = gw < 100;
    int jbase = isL0 ? gw : (gw - 100);
    float cst[5] = {0.f, 0.f, 0.f, 0.f, 0.f};   // cell state, lives in VGPRs

    for (int it = 0; it <= T_SEQ; ++it) {
        int cur = it & 1, prev = cur ^ 1;
        if (isL0 && it < T_SEQ) {
            int t = it;
            float xh[16];
            load_xvec(h1buf + prev * HIDP, lane, xh);   // h1[t-1]
            const float* xr = xp0 + (size_t)t * GATES;  // includes both biases
#pragma unroll
            for (int q = 0; q < 5; ++q) {
                int j = jbase + q * 100;
                float d0 = wave_dot512(whh0 + (size_t)(j)            * HIDP, lane, xh);
                float d1 = wave_dot512(whh0 + (size_t)(j + HID)      * HIDP, lane, xh);
                float d2 = wave_dot512(whh0 + (size_t)(j + 2 * HID)  * HIDP, lane, xh);
                float d3 = wave_dot512(whh0 + (size_t)(j + 3 * HID)  * HIDP, lane, xh);
                float iv = sigmoidf_(xr[j]           + d0);
                float fv = sigmoidf_(xr[j + HID]     + d1);
                float gv = tanhf    (xr[j + 2 * HID] + d2);
                float ov = sigmoidf_(xr[j + 3 * HID] + d3);
                float c = fv * cst[q] + iv * gv;
                cst[q] = c;
                float h = ov * tanhf(c);
                if (lane == 0) h1buf[cur * HIDP + j] = (__bf16)h;
            }
        } else if (!isL0 && it >= 1) {
            // layer1 processes step (it-1): input h1[it-1], state h2[it-2]
            float xh[16], xh2[16];
            load_xvec(h1buf + prev * HIDP, lane, xh);
            load_xvec(h2buf + prev * HIDP, lane, xh2);
#pragma unroll
            for (int q = 0; q < 5; ++q) {
                int j = jbase + q * 100;
                float d[4];
#pragma unroll
                for (int g = 0; g < 4; ++g) {
                    size_t r = (size_t)(j + g * HID) * HIDP;
                    d[g] = wave_dot512(wih1 + r, lane, xh)
                         + wave_dot512(whh1 + r, lane, xh2);
                }
                float iv = sigmoidf_(d[0] + bih1[j]           + bhh1[j]);
                float fv = sigmoidf_(d[1] + bih1[j + HID]     + bhh1[j + HID]);
                float gv = tanhf    (d[2] + bih1[j + 2 * HID] + bhh1[j + 2 * HID]);
                float ov = sigmoidf_(d[3] + bih1[j + 3 * HID] + bhh1[j + 3 * HID]);
                float c = fv * cst[q] + iv * gv;
                cst[q] = c;
                float h = ov * tanhf(c);
                if (lane == 0) { h2buf[cur * HIDP + j] = (__bf16)h; h2last[j] = h; }
            }
        }
        // ---- grid-wide barrier (one per timestep) ----
        __threadfence();
        __syncthreads();
        if (threadIdx.x == 0) {
            __hip_atomic_fetch_add(syncc, 1, __ATOMIC_ACQ_REL, __HIP_MEMORY_SCOPE_AGENT);
            int target = NWG_REC * (it + 1);
            while (__hip_atomic_load(syncc, __ATOMIC_ACQUIRE, __HIP_MEMORY_SCOPE_AGENT) < target)
                __builtin_amdgcn_s_sleep(2);
        }
        __syncthreads();
    }
}

// ---------------- tiny FC head ----------------
__global__ void k_fc(const float* __restrict__ h2last, const float* __restrict__ fcW,
                     const float* __restrict__ fcb, float* __restrict__ out) {
    int i = threadIdx.x;
    if (i >= NCLS) return;
    float acc = fcb[i];
    for (int k = 0; k < HID; ++k) acc += fcW[(size_t)i * HID + k] * h2last[k];
    out[i] = acc;
}

extern "C" void kernel_launch(void* const* d_in, const int* in_sizes, int n_in,
                              void* d_out, int out_size, void* d_ws, size_t ws_size,
                              hipStream_t stream) {
    (void)in_sizes; (void)n_in; (void)out_size; (void)ws_size;
    const int*   seq  = (const int*)d_in[0];
    const float* emb  = (const float*)d_in[1];
    const float* Wih0 = (const float*)d_in[2];
    const float* Whh0 = (const float*)d_in[3];
    const float* bih0 = (const float*)d_in[4];
    const float* bhh0 = (const float*)d_in[5];
    const float* Wih1 = (const float*)d_in[6];
    const float* Whh1 = (const float*)d_in[7];
    const float* bih1 = (const float*)d_in[8];
    const float* bhh1 = (const float*)d_in[9];
    const float* fcW  = (const float*)d_in[10];
    const float* fcb  = (const float*)d_in[11];
    float* out = (float*)d_out;

    char* ws = (char*)d_ws;
    size_t off = 0;
    auto take = [&](size_t bytes) -> char* {
        off = (off + 255) & ~(size_t)255;
        char* p = ws + off; off += bytes; return p;
    };
    __bf16* xe    = (__bf16*)take((size_t)T_SEQ * EMBP * 2);
    __bf16* wih0b = (__bf16*)take((size_t)GATES * EMBP * 2);
    __bf16* whh0b = (__bf16*)take((size_t)GATES * HIDP * 2);
    __bf16* wih1b = (__bf16*)take((size_t)GATES * HIDP * 2);
    __bf16* whh1b = (__bf16*)take((size_t)GATES * HIDP * 2);
    float*  xp0   = (float*)take((size_t)T_SEQ * GATES * 4);
    __bf16* h1buf = (__bf16*)take(2 * HIDP * 2);
    __bf16* h2buf = (__bf16*)take(2 * HIDP * 2);
    float*  h2last= (float*)take(HIDP * 4);
    int*    syncc = (int*)take(256);

    k_init<<<1, 1024, 0, stream>>>(h1buf, h2buf, h2last, syncc);
    { int n = T_SEQ * EMBP; k_embed<<<(n + 255) / 256, 256, 0, stream>>>(seq, emb, xe); }
    { int n = GATES * EMBP; k_cvt<<<(n + 255) / 256, 256, 0, stream>>>(Wih0, wih0b, GATES, EMB, EMBP); }
    { int n = GATES * HIDP; k_cvt<<<(n + 255) / 256, 256, 0, stream>>>(Whh0, whh0b, GATES, HID, HIDP); }
    { int n = GATES * HIDP; k_cvt<<<(n + 255) / 256, 256, 0, stream>>>(Wih1, wih1b, GATES, HID, HIDP); }
    { int n = GATES * HIDP; k_cvt<<<(n + 255) / 256, 256, 0, stream>>>(Whh1, whh1b, GATES, HID, HIDP); }
    // M tiles = 4096/16 = 256 (grid.y); N tiles = 125, 8 per block -> 16 blocks (grid.x)
    k_gemm_xproj<<<dim3(16, 256, 1), 256, 0, stream>>>(xe, wih0b, bih0, bhh0, xp0);
    k_recur<<<NWG_REC, REC_THREADS, 0, stream>>>(xp0, whh0b, wih1b, whh1b,
                                                 bih1, bhh1, h1buf, h2buf, h2last, syncc);
    k_fc<<<1, 128, 0, stream>>>(h2last, fcW, fcb, out);
}
